// MultiHeadSelfAttention_80444737454594
// MI455X (gfx1250) — matrix operand order repaired and checked by reading, not compiled, not run on hardware
//
#include <hip/hip_runtime.h>
#include <hip/hip_bf16.h>

#define BATCH   4
#define SEQ     1024
#define DMODEL  512
#define NHEAD   8
#define HDIM    64
#define RELROWS 2049
#define ATT_SCALE 0.04419417382415922f   // 512^-0.5
#define NEG_INF_F (-1e9f)

typedef __attribute__((ext_vector_type(16))) _Float16 v16h;
typedef __attribute__((ext_vector_type(8)))  _Float16 v8h;
typedef __attribute__((ext_vector_type(8)))  float    v8f;

// Build a v16h WMMA fragment from two contiguous 8-half (16B) runs.
__device__ __forceinline__ v16h ldfrag(const _Float16* p0, const _Float16* p1) {
  v8h a = *(const v8h*)p0;
  v8h b = *(const v8h*)p1;
  v16h r;
#pragma unroll
  for (int i = 0; i < 8; ++i) { r[i] = a[i]; r[i + 8] = b[i]; }
  return r;
}

__device__ __forceinline__ v8f wmma_f16(v16h a, v16h b, v8f c) {
  // D = A(16x32,f16) * B(32x16,f16) + C(16x16,f32)
  return __builtin_amdgcn_wmma_f32_16x16x32_f16(false, a, false, b, (short)0, c,
                                                false, false);
}

// ---------------------------------------------------------------- prep kernels
__global__ __launch_bounds__(256) void cvt_f32_to_f16(const float* __restrict__ in,
                                                      _Float16* __restrict__ out, int n) {
  int i = blockIdx.x * 256 + threadIdx.x;
  if (i < n) out[i] = (_Float16)in[i];
}

// in[K][N] (f32) -> out[N][K] (f16)
__global__ __launch_bounds__(256) void transpose_cvt(const float* __restrict__ in,
                                                     _Float16* __restrict__ out,
                                                     int K, int N) {
  int i = blockIdx.x * 256 + threadIdx.x;
  if (i < K * N) {
    int k = i / N, n = i - k * N;
    out[(size_t)n * K + k] = (_Float16)in[i];
  }
}

// ---------------------------------------------------- QKV projection (WMMA GEMM)
// C[4096][1536] = X16[4096][512] * WqkvT[1536][512]^T, scattered to q/k/vt ws.
// Register-blocked: each wave computes a 32x64 strip (2 M-tiles x 4 N-tiles),
// so each k-chunk issues 6 fragment loads for 8 WMMAs (A reused 4x, B reused 2x).
__global__ __launch_bounds__(128) void gemm_qkv(const _Float16* __restrict__ A,
                                                const _Float16* __restrict__ Bt,
                                                _Float16* __restrict__ q16,
                                                _Float16* __restrict__ k16,
                                                _Float16* __restrict__ vt16) {
  const int lane = threadIdx.x & 31;
  const int wave = threadIdx.x >> 5;
  const int ln   = lane & 15;
  const int b8a  = (lane < 16) ? 0 : 8;    // accumulator row base per lane half
  const int o16  = (lane < 16) ? 0 : 16;   // B-fragment K-base per lane half
  const int m0 = blockIdx.x * 32;
  const int n0 = blockIdx.y * 256 + wave * 64;

  const _Float16* arow0 = A + (size_t)(m0 + ln) * DMODEL;
  const _Float16* arow1 = A + (size_t)(m0 + 16 + ln) * DMODEL;
  const _Float16* brow0 = Bt + (size_t)(n0 + ln) * DMODEL;
  const _Float16* brow1 = Bt + (size_t)(n0 + 16 + ln) * DMODEL;
  const _Float16* brow2 = Bt + (size_t)(n0 + 32 + ln) * DMODEL;
  const _Float16* brow3 = Bt + (size_t)(n0 + 48 + ln) * DMODEL;

  v8f c[2][4] = {};
  for (int k = 0; k < DMODEL; k += 32) {
    v16h a0 = ldfrag(arow0 + k + o16, arow0 + k + o16 + 8);
    v16h a1 = ldfrag(arow1 + k + o16, arow1 + k + o16 + 8);
    v16h b0 = ldfrag(brow0 + k + o16, brow0 + k + o16 + 8);
    v16h b1 = ldfrag(brow1 + k + o16, brow1 + k + o16 + 8);
    v16h b2 = ldfrag(brow2 + k + o16, brow2 + k + o16 + 8);
    v16h b3 = ldfrag(brow3 + k + o16, brow3 + k + o16 + 8);
    c[0][0] = wmma_f16(a0, b0, c[0][0]);
    c[0][1] = wmma_f16(a0, b1, c[0][1]);
    c[0][2] = wmma_f16(a0, b2, c[0][2]);
    c[0][3] = wmma_f16(a0, b3, c[0][3]);
    c[1][0] = wmma_f16(a1, b0, c[1][0]);
    c[1][1] = wmma_f16(a1, b1, c[1][1]);
    c[1][2] = wmma_f16(a1, b2, c[1][2]);
    c[1][3] = wmma_f16(a1, b3, c[1][3]);
  }
#pragma unroll
  for (int mi = 0; mi < 2; ++mi) {
#pragma unroll
    for (int j = 0; j < 4; ++j) {
#pragma unroll
      for (int v = 0; v < 8; ++v) {
        int M = v + b8a;
        int m = m0 + mi * 16 + M;       // token index
        int col = n0 + j * 16 + ln;     // 0..1535
        int which = col >> 9;           // 0:q 1:k 2:v
        int rem = col & 511;
        int h = rem >> 6, d = rem & 63;
        int b = m >> 10, s = m & 1023;
        _Float16 val = (_Float16)c[mi][j][v];
        size_t bh = (size_t)(b * NHEAD + h);
        if (which == 0)      q16[(bh * SEQ + s) * HDIM + d] = val;
        else if (which == 1) k16[(bh * SEQ + s) * HDIM + d] = val;
        else                 vt16[(bh * HDIM + d) * SEQ + s] = val;  // V^T
      }
    }
  }
}

// ----------------------------------------------------------- attention (flash)
// One wave32 per (b*h, 16-row block). WMMA for QK^T, rel-pos band, and PV.
__global__ __launch_bounds__(32) void attn_kernel(const _Float16* __restrict__ q16,
                                                  const _Float16* __restrict__ k16,
                                                  const _Float16* __restrict__ vt16,
                                                  const _Float16* __restrict__ rel16,
                                                  const unsigned char* __restrict__ mask,
                                                  _Float16* __restrict__ ao16) {
  __shared__ float    st[16][32];     // masked+scaled scores
  __shared__ float    gc[16][48];     // rel-pos band G[n][j]
  __shared__ _Float16 pt[16][32];     // probabilities (A-fragment source)
  __shared__ float    scs[16];        // per-row rescale factors
  __shared__ float    rls[16];        // per-row 1/l

  const int lane = threadIdx.x;
  const int ln   = lane & 15;
  const int hi   = lane >> 4;         // 0 or 1 (lane half)
  const int b8a  = hi ? 8 : 0;
  const int o16  = hi ? 16 : 0;
  const int bh = blockIdx.y;          // 0..31
  const int n0 = blockIdx.x * 16;
  const int b  = bh >> 3;
  const int h  = bh & 7;

  const _Float16* qbase  = q16  + (size_t)bh * SEQ * HDIM;
  const _Float16* kbase  = k16  + (size_t)bh * SEQ * HDIM;
  const _Float16* vtbase = vt16 + (size_t)bh * HDIM * SEQ;

  // Q A-fragments (held for whole row block): K-chunks 0..31 and 32..63
  const _Float16* qrow = qbase + (size_t)(n0 + ln) * HDIM;
  v16h qa0 = ldfrag(qrow + o16,      qrow + o16 + 8);
  v16h qa1 = ldfrag(qrow + 32 + o16, qrow + 32 + o16 + 8);

  v8f o[4] = {v8f{}, v8f{}, v8f{}, v8f{}};
  float m_i = -INFINITY, l_i = 0.f;

  for (int r0 = 0; r0 < SEQ; r0 += 32) {
    if (r0 + 32 < SEQ) {   // prefetch next K/V^T tiles (global_prefetch_b8)
      __builtin_prefetch(kbase + (size_t)(r0 + 32 + ln) * HDIM, 0, 1);
      __builtin_prefetch(vtbase + (size_t)ln * SEQ + r0 + 32, 0, 1);
    }
    // ---- rel-pos band: G[n][j] = q[n] . rel_emb[lo + j], j in [0,48)
    const int lo = n0 - r0 - 31 + 1024;   // in [1, 2001]; lo+47 <= 2048
#pragma unroll
    for (int j0i = 0; j0i < 3; ++j0i) {
      const int j0 = j0i * 16;
      const _Float16* rrow = rel16 + (size_t)(lo + j0 + ln) * HDIM;
      v16h rb0 = ldfrag(rrow + o16,      rrow + o16 + 8);
      v16h rb1 = ldfrag(rrow + 32 + o16, rrow + 32 + o16 + 8);
      v8f g = {};
      g = wmma_f16(qa0, rb0, g);
      g = wmma_f16(qa1, rb1, g);
#pragma unroll
      for (int v = 0; v < 8; ++v) gc[v + b8a][j0 + ln] = g[v];
    }
    // ---- scores: two 16x16 tiles over this 32-column block
    v8f sf[2];
#pragma unroll
    for (int t = 0; t < 2; ++t) {
      const _Float16* krow = kbase + (size_t)(r0 + t * 16 + ln) * HDIM;
      v16h kb0 = ldfrag(krow + o16,      krow + o16 + 8);
      v16h kb1 = ldfrag(krow + 32 + o16, krow + 32 + o16 + 8);
      v8f s = {};
      s = wmma_f16(qa0, kb0, s);
      s = wmma_f16(qa1, kb1, s);
      sf[t] = s;
    }
    __syncthreads();   // gc visible
    // ---- add pos (anti-diagonal gather from band), scale, mask
#pragma unroll
    for (int t = 0; t < 2; ++t) {
      const int r = r0 + t * 16 + ln;
      const bool mk = mask[b * SEQ + r] != 0;
      const int shift = (t == 0) ? 31 : 15;
#pragma unroll
      for (int v = 0; v < 8; ++v) {
        int M = v + b8a;
        float s = (sf[t][v] + gc[M][M - ln + shift]) * ATT_SCALE;
        st[M][t * 16 + ln] = mk ? s : NEG_INF_F;
      }
    }
    __syncthreads();
    // ---- online softmax: lane pair (ln, ln+16) owns row ln (16 cols each)
    {
      float pm = -INFINITY;
#pragma unroll
      for (int c = 0; c < 16; ++c) pm = fmaxf(pm, st[ln][hi * 16 + c]);
      pm = fmaxf(pm, __shfl_xor(pm, 16, 32));
      float nm = fmaxf(m_i, pm);
      float ps = 0.f;
#pragma unroll
      for (int c = 0; c < 16; ++c) {
        float p = __expf(st[ln][hi * 16 + c] - nm);
        ps += p;
        pt[ln][hi * 16 + c] = (_Float16)p;
      }
      ps += __shfl_xor(ps, 16, 32);
      float sc = __expf(m_i - nm);
      l_i = l_i * sc + ps;
      m_i = nm;
      if (lane < 16) scs[ln] = sc;
    }
    __syncthreads();
    // ---- rescale accumulators, then PV via WMMA
#pragma unroll
    for (int v = 0; v < 8; ++v) {
      float sc = scs[v + b8a];
      o[0][v] *= sc; o[1][v] *= sc; o[2][v] *= sc; o[3][v] *= sc;
    }
    const _Float16* prow = &pt[ln][0];
    v16h pa = ldfrag(prow + o16, prow + o16 + 8);   // A = P (16 x 32)
#pragma unroll
    for (int d0i = 0; d0i < 4; ++d0i) {
      const _Float16* vrow = vtbase + (size_t)(d0i * 16 + ln) * SEQ + r0 + o16;
      v16h vb = ldfrag(vrow, vrow + 8);              // B = V (32 x 16), from V^T
      o[d0i] = wmma_f16(pa, vb, o[d0i]);
    }
    __syncthreads();   // protect st/gc/pt before next block
  }
  if (lane < 16) rls[ln] = 1.0f / l_i;
  __syncthreads();
  // ---- store normalized output, token-major [B*S][H*hd] (f16)
#pragma unroll
  for (int d0i = 0; d0i < 4; ++d0i) {
#pragma unroll
    for (int v = 0; v < 8; ++v) {
      int M = v + b8a;
      int n = n0 + M;
      int d = d0i * 16 + ln;
      ao16[(size_t)(b * SEQ + n) * DMODEL + h * HDIM + d] =
          (_Float16)(o[d0i][v] * rls[M]);
    }
  }
}

// ------------------------------------------------- output projection (WMMA GEMM)
// Register-blocked 32x64 per wave, writes f32 directly to d_out.
__global__ __launch_bounds__(128) void gemm_out(const _Float16* __restrict__ A,
                                                const _Float16* __restrict__ Bt,
                                                float* __restrict__ out) {
  const int lane = threadIdx.x & 31;
  const int wave = threadIdx.x >> 5;
  const int ln   = lane & 15;
  const int b8a  = (lane < 16) ? 0 : 8;
  const int o16  = (lane < 16) ? 0 : 16;
  const int m0 = blockIdx.x * 32;
  const int n0 = blockIdx.y * 256 + wave * 64;

  const _Float16* arow0 = A + (size_t)(m0 + ln) * DMODEL;
  const _Float16* arow1 = A + (size_t)(m0 + 16 + ln) * DMODEL;
  const _Float16* brow0 = Bt + (size_t)(n0 + ln) * DMODEL;
  const _Float16* brow1 = Bt + (size_t)(n0 + 16 + ln) * DMODEL;
  const _Float16* brow2 = Bt + (size_t)(n0 + 32 + ln) * DMODEL;
  const _Float16* brow3 = Bt + (size_t)(n0 + 48 + ln) * DMODEL;

  v8f c[2][4] = {};
  for (int k = 0; k < DMODEL; k += 32) {
    v16h a0 = ldfrag(arow0 + k + o16, arow0 + k + o16 + 8);
    v16h a1 = ldfrag(arow1 + k + o16, arow1 + k + o16 + 8);
    v16h b0 = ldfrag(brow0 + k + o16, brow0 + k + o16 + 8);
    v16h b1 = ldfrag(brow1 + k + o16, brow1 + k + o16 + 8);
    v16h b2 = ldfrag(brow2 + k + o16, brow2 + k + o16 + 8);
    v16h b3 = ldfrag(brow3 + k + o16, brow3 + k + o16 + 8);
    c[0][0] = wmma_f16(a0, b0, c[0][0]);
    c[0][1] = wmma_f16(a0, b1, c[0][1]);
    c[0][2] = wmma_f16(a0, b2, c[0][2]);
    c[0][3] = wmma_f16(a0, b3, c[0][3]);
    c[1][0] = wmma_f16(a1, b0, c[1][0]);
    c[1][1] = wmma_f16(a1, b1, c[1][1]);
    c[1][2] = wmma_f16(a1, b2, c[1][2]);
    c[1][3] = wmma_f16(a1, b3, c[1][3]);
  }
#pragma unroll
  for (int mi = 0; mi < 2; ++mi) {
#pragma unroll
    for (int j = 0; j < 4; ++j) {
#pragma unroll
      for (int v = 0; v < 8; ++v) {
        int M = v + b8a;
        out[(size_t)(m0 + mi * 16 + M) * DMODEL + n0 + j * 16 + ln] = c[mi][j][v];
      }
    }
  }
}

// ---------------------------------------------------------------------- launch
extern "C" void kernel_launch(void* const* d_in, const int* in_sizes, int n_in,
                              void* d_out, int out_size, void* d_ws, size_t ws_size,
                              hipStream_t stream) {
  const float* x    = (const float*)d_in[0];             // [4,1024,512]
  const unsigned char* mask = (const unsigned char*)d_in[1]; // [4,1024] bool
  const float* Wqkv = (const float*)d_in[2];             // [512,1536]
  const float* Wout = (const float*)d_in[3];             // [512,512]
  const float* rel  = (const float*)d_in[4];             // [2049,64]
  float* out = (float*)d_out;

  size_t off = 0;
  auto carve = [&](size_t elems) -> _Float16* {
    _Float16* p = (_Float16*)((char*)d_ws + off);
    off += (elems * sizeof(_Float16) + 255) & ~(size_t)255;
    return p;
  };
  const size_t NTOK = (size_t)BATCH * SEQ;                   // 4096
  _Float16* x16    = carve(NTOK * DMODEL);                   // [4096][512]
  _Float16* wqkvT  = carve((size_t)3 * DMODEL * DMODEL);     // [1536][512]
  _Float16* woutT  = carve((size_t)DMODEL * DMODEL);         // [512][512]
  _Float16* rel16  = carve((size_t)RELROWS * HDIM);          // [2049][64]
  _Float16* q16    = carve((size_t)BATCH * NHEAD * SEQ * HDIM);
  _Float16* k16    = carve((size_t)BATCH * NHEAD * SEQ * HDIM);
  _Float16* vt16   = carve((size_t)BATCH * NHEAD * HDIM * SEQ);
  _Float16* ao16   = carve(NTOK * DMODEL);                   // attention output

  // 1) precision conversion / weight transposes
  {
    int n = (int)(NTOK * DMODEL);
    cvt_f32_to_f16<<<(n + 255) / 256, 256, 0, stream>>>(x, x16, n);
  }
  {
    int n = DMODEL * 3 * DMODEL;
    transpose_cvt<<<(n + 255) / 256, 256, 0, stream>>>(Wqkv, wqkvT, DMODEL, 3 * DMODEL);
  }
  {
    int n = DMODEL * DMODEL;
    transpose_cvt<<<(n + 255) / 256, 256, 0, stream>>>(Wout, woutT, DMODEL, DMODEL);
  }
  {
    int n = RELROWS * HDIM;
    cvt_f32_to_f16<<<(n + 255) / 256, 256, 0, stream>>>(rel, rel16, n);
  }
  // 2) QKV projection (32x64 strip per wave, 8 WMMAs per k-chunk)
  gemm_qkv<<<dim3(NTOK / 32, (3 * DMODEL) / 256), 128, 0, stream>>>(
      x16, wqkvT, q16, k16, vt16);
  // 3) attention with relative-position bias (flash, WMMA)
  attn_kernel<<<dim3(SEQ / 16, BATCH * NHEAD), 32, 0, stream>>>(
      q16, k16, vt16, rel16, mask, ao16);
  // 4) output projection -> f32 d_out
  gemm_out<<<dim3(NTOK / 32, DMODEL / 256), 128, 0, stream>>>(ao16, woutT, out);
}
